// CrossAttentionFusion_54941221650613
// MI455X (gfx1250) — compile-verified, matrix operand order checked by reading, not run on hardware
//
#include <hip/hip_runtime.h>
#include <hip/hip_bf16.h>

// ---------------------------------------------------------------------------
// CrossAttentionFusion for MI455X (gfx1250, wave32, WMMA bf16)
// B=8, S=2048, D=512, HID=256
// ---------------------------------------------------------------------------

#define B_  8
#define S_  2048
#define D_  512
#define HID_ 256
#define BS_ (B_ * S_)
#define OUTW_ (3 * D_)

typedef __bf16 v16bf __attribute__((ext_vector_type(16)));
typedef float  v8f   __attribute__((ext_vector_type(8)));

union Frag16 {
    v16bf v;
    uint4 q[2];
    unsigned short h[16];
};

union Acc8 {
    v8f v;
    float f[8];
};

__device__ __forceinline__ unsigned short f2bf(float x) {
    unsigned int u = __float_as_uint(x);
    u += 0x7FFFu + ((u >> 16) & 1u);   // round-to-nearest-even
    return (unsigned short)(u >> 16);
}
__device__ __forceinline__ float bf2f(unsigned short h) {
    return __uint_as_float(((unsigned int)h) << 16);
}
__device__ __forceinline__ float sigmoidf_(float x) {
    return 1.0f / (1.0f + __expf(-x));
}

// ---------------------------------------------------------------------------
// Generic GEMM: out[m][n] = act( A[m][:] @ W[:][n] + bias[n] ), bf16 output.
// A: f32 [BS_ x K], W: f32 [K x N] row-major, out: bf16.
// Block: 256 threads (8 waves). Block tile: 128 rows x 64 cols.
// Wave w handles rows [16w, 16w+16), all 64 cols (4 WMMA accum tiles).
// W tile is staged transposed into LDS (bf16) so B-fragments are b128 loads.
// transposedOut: write out as [b][n][s] (for V, so P@V B-frags are contiguous).
// ---------------------------------------------------------------------------
__global__ __launch_bounds__(256)
void caf_gemm_bf16(const float* __restrict__ A, const float* __restrict__ W,
                   const float* __restrict__ bias, unsigned short* __restrict__ out,
                   int K, int N, int nTiles, int doRelu, int transposedOut) {
    __shared__ __align__(16) unsigned short Wt[64][40];   // [n][k], padded row

    const int mt = blockIdx.x / nTiles;
    const int nt = blockIdx.x % nTiles;
    const int m0 = mt * 128;
    const int n0 = nt * 64;

    const int w    = threadIdx.x >> 5;
    const int lane = threadIdx.x & 31;
    const int klo  = 8 * (lane >> 4);
    const int rowA = m0 + w * 16 + (lane & 15);

    Acc8 acc[4];
    #pragma unroll
    for (int n = 0; n < 4; ++n)
        #pragma unroll
        for (int e = 0; e < 8; ++e) acc[n].f[e] = 0.f;

    for (int kk = 0; kk < K; kk += 32) {
        __syncthreads();
        // Stage W[kk..kk+32) x [n0..n0+64) transposed into LDS as bf16.
        // 32 * 16 = 512 float4 chunks; 2 per thread.
        #pragma unroll
        for (int it = 0; it < 2; ++it) {
            int i  = threadIdx.x * 2 + it;
            int k  = i >> 4;
            int n4 = (i & 15) * 4;
            float4 wv = *(const float4*)&W[(size_t)(kk + k) * N + n0 + n4];
            Wt[n4 + 0][k] = f2bf(wv.x);
            Wt[n4 + 1][k] = f2bf(wv.y);
            Wt[n4 + 2][k] = f2bf(wv.z);
            Wt[n4 + 3][k] = f2bf(wv.w);
        }
        __syncthreads();

        // A fragment: rows m0+16w+(lane&15), K runs [klo..klo+7], [16+klo..+23]
        Frag16 af;
        {
            const float* ap = &A[(size_t)rowA * K + kk + klo];
            float4 x0 = *(const float4*)(ap + 0);
            float4 x1 = *(const float4*)(ap + 4);
            float4 x2 = *(const float4*)(ap + 16);
            float4 x3 = *(const float4*)(ap + 20);
            af.h[0] = f2bf(x0.x); af.h[1] = f2bf(x0.y); af.h[2] = f2bf(x0.z); af.h[3] = f2bf(x0.w);
            af.h[4] = f2bf(x1.x); af.h[5] = f2bf(x1.y); af.h[6] = f2bf(x1.z); af.h[7] = f2bf(x1.w);
            af.h[8]  = f2bf(x2.x); af.h[9]  = f2bf(x2.y); af.h[10] = f2bf(x2.z); af.h[11] = f2bf(x2.w);
            af.h[12] = f2bf(x3.x); af.h[13] = f2bf(x3.y); af.h[14] = f2bf(x3.z); af.h[15] = f2bf(x3.w);
        }

        #pragma unroll
        for (int n = 0; n < 4; ++n) {
            Frag16 bfr;
            int col = n * 16 + (lane & 15);
            bfr.q[0] = *(const uint4*)&Wt[col][klo];
            bfr.q[1] = *(const uint4*)&Wt[col][klo + 16];
            acc[n].v = __builtin_amdgcn_wmma_f32_16x16x32_bf16(
                false, af.v, false, bfr.v, (short)0, acc[n].v, false, false);
        }
    }

    // Epilogue: bias, activation, bf16 store
    #pragma unroll
    for (int n = 0; n < 4; ++n) {
        int colg = n0 + n * 16 + (lane & 15);
        float bv = bias[colg];
        #pragma unroll
        for (int e = 0; e < 8; ++e) {
            int rowg = m0 + w * 16 + e + 8 * (lane >> 4);
            float r = acc[n].f[e] + bv;
            if (doRelu) r = fmaxf(r, 0.f);
            unsigned short hv = f2bf(r);
            if (transposedOut) {
                int b = rowg >> 11;          // /S_
                int s = rowg & (S_ - 1);
                out[((size_t)b * D_ + colg) * S_ + s] = hv;
            } else {
                out[(size_t)rowg * N + colg] = hv;
            }
        }
    }
}

// ---------------------------------------------------------------------------
// Flash attention + fused_l epilogue.
// Grid: B * S/32 blocks of 256 threads (8 waves).
// Wave w: rowgroup rg=w/4 (16 q rows), D-slice c=w%4 (128 cols of O).
// Per key tile (32 keys): partial QK^T via WMMA -> ds_add_f32 reduce in LDS ->
// online softmax by threads 0..31 -> P bf16 in LDS -> O = O*alpha + P@Vt.
// ---------------------------------------------------------------------------
__global__ __launch_bounds__(256)
void caf_flash_attn(const unsigned short* __restrict__ qb,   // [B][S][D] bf16
                    const unsigned short* __restrict__ kb,   // [B][S][D] bf16
                    const unsigned short* __restrict__ vtb,  // [B][D][S] bf16
                    const float* __restrict__ lin,           // [B][S][D] f32
                    const float* __restrict__ alpha_l,
                    float* __restrict__ out) {
    __shared__ __align__(16) float Sred[32][33];
    __shared__ __align__(16) unsigned short Pt[32][32];
    __shared__ float mSt[32], lSt[32], aSt[32];

    const int b  = blockIdx.x >> 6;
    const int q0 = (blockIdx.x & 63) * 32;

    const int w    = threadIdx.x >> 5;
    const int lane = threadIdx.x & 31;
    const int rg   = w >> 2;
    const int c    = w & 3;
    const int klo  = 8 * (lane >> 4);
    const float scale = 0.04419417382415922f;   // 1/sqrt(512)
    const float gl = sigmoidf_(alpha_l[0]);

    if (threadIdx.x < 32) { mSt[threadIdx.x] = -1e30f; lSt[threadIdx.x] = 0.f; }

    // Preload Q fragments for this wave's 16 rows x 128-col D slice (4 k-steps)
    Frag16 qa[4];
    {
        const int qrow = q0 + rg * 16 + (lane & 15);
        const unsigned short* qp = &qb[((size_t)b * S_ + qrow) * D_];
        #pragma unroll
        for (int k = 0; k < 4; ++k) {
            int d0 = 128 * c + 32 * k + klo;
            qa[k].q[0] = *(const uint4*)&qp[d0];
            qa[k].q[1] = *(const uint4*)&qp[d0 + 16];
        }
    }

    Acc8 O[8];
    #pragma unroll
    for (int n = 0; n < 8; ++n)
        #pragma unroll
        for (int e = 0; e < 8; ++e) O[n].f[e] = 0.f;

    for (int j = 0; j < S_ / 32; ++j) {
        // -- zero S reduction buffer --
        __syncthreads();
        for (int i = threadIdx.x; i < 32 * 33; i += 256) (&Sred[0][0])[i] = 0.f;
        __syncthreads();

        // -- partial scores: S[16rg.., 32 keys] over d in [128c, 128c+128) --
        #pragma unroll
        for (int n = 0; n < 2; ++n) {
            Acc8 sa;
            #pragma unroll
            for (int e = 0; e < 8; ++e) sa.f[e] = 0.f;
            int key = j * 32 + n * 16 + (lane & 15);
            const unsigned short* kp = &kb[((size_t)b * S_ + key) * D_];
            #pragma unroll
            for (int k = 0; k < 4; ++k) {
                Frag16 bfr;
                int d0 = 128 * c + 32 * k + klo;
                bfr.q[0] = *(const uint4*)&kp[d0];
                bfr.q[1] = *(const uint4*)&kp[d0 + 16];
                sa.v = __builtin_amdgcn_wmma_f32_16x16x32_bf16(
                    false, qa[k].v, false, bfr.v, (short)0, sa.v, false, false);
            }
            int col = n * 16 + (lane & 15);
            #pragma unroll
            for (int e = 0; e < 8; ++e) {
                int row = rg * 16 + e + 8 * (lane >> 4);
                atomicAdd(&Sred[row][col], sa.f[e]);
            }
        }
        __syncthreads();

        // -- online softmax (one thread per query row) --
        if (threadIdx.x < 32) {
            int t = threadIdx.x;
            float m_old = mSt[t];
            float mx = m_old;
            float srow[32];
            #pragma unroll
            for (int jj = 0; jj < 32; ++jj) {
                float s = Sred[t][jj] * scale;
                srow[jj] = s;
                mx = fmaxf(mx, s);
            }
            float al = __expf(m_old - mx);
            float lsum = 0.f;
            #pragma unroll
            for (int jj = 0; jj < 32; ++jj) {
                float p = __expf(srow[jj] - mx);
                lsum += p;
                Pt[t][jj] = f2bf(p);
            }
            mSt[t] = mx;
            lSt[t] = lSt[t] * al + lsum;
            aSt[t] = al;
        }
        __syncthreads();

        // -- O = O * alpha + P @ Vt --
        float a8[8];
        #pragma unroll
        for (int e = 0; e < 8; ++e) a8[e] = aSt[rg * 16 + e + 8 * (lane >> 4)];

        Frag16 pf;
        {
            int prow = rg * 16 + (lane & 15);
            pf.q[0] = *(const uint4*)&Pt[prow][klo];
            pf.q[1] = *(const uint4*)&Pt[prow][klo + 16];
        }
        #pragma unroll
        for (int n = 0; n < 8; ++n) {
            #pragma unroll
            for (int e = 0; e < 8; ++e) O[n].f[e] *= a8[e];
            Frag16 vf;
            int d = 128 * c + 16 * n + (lane & 15);
            const unsigned short* vp = &vtb[((size_t)b * D_ + d) * S_ + j * 32];
            vf.q[0] = *(const uint4*)&vp[klo];
            vf.q[1] = *(const uint4*)&vp[klo + 16];
            O[n].v = __builtin_amdgcn_wmma_f32_16x16x32_bf16(
                false, pf.v, false, vf.v, (short)0, O[n].v, false, false);
        }
    }
    __syncthreads();

    // -- epilogue: fused_l = gl * O/l + (2-gl) * l_in --
    float linv[8];
    #pragma unroll
    for (int e = 0; e < 8; ++e) linv[e] = 1.0f / lSt[rg * 16 + e + 8 * (lane >> 4)];

    #pragma unroll
    for (int n = 0; n < 8; ++n) {
        int d = 128 * c + 16 * n + (lane & 15);
        #pragma unroll
        for (int e = 0; e < 8; ++e) {
            int srow = q0 + rg * 16 + e + 8 * (lane >> 4);
            float li = lin[((size_t)b * S_ + srow) * D_ + d];
            out[((size_t)b * S_ + srow) * OUTW_ + d] = gl * O[n].f[e] * linv[e] + (2.f - gl) * li;
        }
    }
}

// ---------------------------------------------------------------------------
// Per-row h @ W2 + b2 -> sigmoid. One wave per row (8 rows / block).
// ---------------------------------------------------------------------------
__global__ __launch_bounds__(256)
void caf_vgate_dot(const unsigned short* __restrict__ hb, const float* __restrict__ W2,
                   const float* __restrict__ b2, float* __restrict__ vw) {
    const int w    = threadIdx.x >> 5;
    const int lane = threadIdx.x & 31;
    const int row  = blockIdx.x * 8 + w;

    Frag16 hx;
    hx.q[0] = *(const uint4*)&hb[(size_t)row * HID_ + lane * 8];
    float sum = 0.f;
    #pragma unroll
    for (int i = 0; i < 8; ++i)
        sum += bf2f(hx.h[i]) * W2[lane * 8 + i];
    #pragma unroll
    for (int off = 16; off > 0; off >>= 1)
        sum += __shfl_down(sum, off, 32);
    if (lane == 0) vw[row] = sigmoidf_(sum + b2[0]);
}

// ---------------------------------------------------------------------------
// Elementwise: fused_a = (1+sigmoid(alpha_a))*a ; fused_v = vw*v
// ---------------------------------------------------------------------------
__global__ __launch_bounds__(256)
void caf_fuse_av(const float* __restrict__ a, const float* __restrict__ v,
                 const float* __restrict__ vw, const float* __restrict__ alpha_a,
                 float* __restrict__ out) {
    size_t idx = (size_t)blockIdx.x * 256 + threadIdx.x;   // over B*S*D
    size_t row = idx >> 9;           // /D_
    int    d   = (int)(idx & (D_ - 1));
    float ga = sigmoidf_(alpha_a[0]);
    out[row * OUTW_ + D_ + d]     = (1.f + ga) * a[idx];
    out[row * OUTW_ + 2 * D_ + d] = vw[row] * v[idx];
}

// ---------------------------------------------------------------------------
extern "C" void kernel_launch(void* const* d_in, const int* in_sizes, int n_in,
                              void* d_out, int out_size, void* d_ws, size_t ws_size,
                              hipStream_t stream) {
    (void)in_sizes; (void)n_in; (void)out_size; (void)ws_size;
    const float* a       = (const float*)d_in[0];
    const float* v       = (const float*)d_in[1];
    const float* l       = (const float*)d_in[2];
    const float* Wq      = (const float*)d_in[3];
    const float* bq      = (const float*)d_in[4];
    const float* Wk      = (const float*)d_in[5];
    const float* bk      = (const float*)d_in[6];
    const float* Wv      = (const float*)d_in[7];
    const float* bv      = (const float*)d_in[8];
    const float* W1      = (const float*)d_in[9];
    const float* b1      = (const float*)d_in[10];
    const float* W2      = (const float*)d_in[11];
    const float* b2      = (const float*)d_in[12];
    const float* alpha_a = (const float*)d_in[13];
    const float* alpha_l = (const float*)d_in[14];
    float* out = (float*)d_out;

    char* ws = (char*)d_ws;
    const size_t MB = 1024u * 1024u;
    unsigned short* qb  = (unsigned short*)(ws + 0 * MB);
    unsigned short* kb  = (unsigned short*)(ws + 16 * MB);
    unsigned short* vtb = (unsigned short*)(ws + 32 * MB);
    unsigned short* hb  = (unsigned short*)(ws + 48 * MB);
    float*          vw  = (float*)(ws + 56 * MB);

    dim3 blk(256);

    // Projections (K=512). grid = (BS/128) * (N/64)
    caf_gemm_bf16<<<dim3(128 * 8), blk, 0, stream>>>(l, Wq, bq, qb,  D_, D_,   8, 0, 0);
    caf_gemm_bf16<<<dim3(128 * 8), blk, 0, stream>>>(a, Wk, bk, kb,  D_, D_,   8, 0, 0);
    caf_gemm_bf16<<<dim3(128 * 8), blk, 0, stream>>>(a, Wv, bv, vtb, D_, D_,   8, 0, 1);
    caf_gemm_bf16<<<dim3(128 * 4), blk, 0, stream>>>(v, W1, b1, hb,  D_, HID_, 4, 1, 0);

    // Visual gate per-row dot + sigmoid
    caf_vgate_dot<<<dim3(BS_ / 8), blk, 0, stream>>>(hb, W2, b2, vw);

    // Flash attention + fused_l epilogue
    caf_flash_attn<<<dim3(B_ * (S_ / 32)), blk, 0, stream>>>(qb, kb, vtb, l, alpha_l, out);

    // fused_a, fused_v
    caf_fuse_av<<<dim3((BS_ * D_) / 256), blk, 0, stream>>>(a, v, vw, alpha_a, out);
}
